// LSTMTagger_26104811225778
// MI455X (gfx1250) — compile-verified, matrix operand order checked by reading, not run on hardware
//
#include <hip/hip_runtime.h>

// ---------------------------------------------------------------------------
// Problem constants (must match reference)
// ---------------------------------------------------------------------------
constexpr int kB  = 8;
constexpr int kL  = 2048;
constexpr int kD  = 1024;
constexpr int kNL = 16;
constexpr int kH  = 64;
constexpr int kE  = 16;     // kD / kH
constexpr int kS  = 256;    // kD / 4
constexpr int kCAT = kNL * kS + kD;   // 5120
constexpr float kEps = 1e-5f;
constexpr float kInvSqrtL = 0.022097086912079608f;  // 1/sqrt(2048)

// ---------------------------------------------------------------------------
// CDNA5 WMMA types
// ---------------------------------------------------------------------------
typedef __attribute__((ext_vector_type(16))) __bf16 v16bf;
typedef __attribute__((ext_vector_type(8)))  float  v8f;

union Frag {
    uint4          q[2];
    unsigned short s[16];
    v16bf          v;
};

// Async global->LDS copy path (CDNA5): gated on builtin availability.
// Probe-learned prototype: param0 = AS1 pointer to gcc-vector int4 (non-const),
// param1 = AS3 pointer, then imm offset + imm cpol.
#if __has_builtin(__builtin_amdgcn_global_load_async_to_lds_b128)
#define USE_ASYNC_LDS 1
typedef int v4i_t __attribute__((vector_size(16)));
typedef __attribute__((address_space(1))) v4i_t* gv4_p;
typedef __attribute__((address_space(3))) v4i_t* lv4_p;
#endif

__device__ __forceinline__ void wait_async_lds() {
#if __has_builtin(__builtin_amdgcn_s_wait_asynccnt)
    __builtin_amdgcn_s_wait_asynccnt(0);
#else
    asm volatile("s_wait_asynccnt 0x0" ::: "memory");
#endif
}

__device__ __forceinline__ unsigned short f2bf(float x) {
    unsigned int u = __float_as_uint(x);
    u += 0x7FFFu + ((u >> 16) & 1u);   // round-to-nearest-even
    return (unsigned short)(u >> 16);
}

// block-wide reductions; blockDim.x == 256 assumed
__device__ __forceinline__ float blockReduceSum(float v, float* sm) {
    int t = threadIdx.x;
    sm[t] = v; __syncthreads();
    #pragma unroll
    for (int s = 128; s > 0; s >>= 1) {
        if (t < s) sm[t] += sm[t + s];
        __syncthreads();
    }
    float r = sm[0]; __syncthreads();
    return r;
}
__device__ __forceinline__ float blockReduceMax(float v, float* sm) {
    int t = threadIdx.x;
    sm[t] = v; __syncthreads();
    #pragma unroll
    for (int s = 128; s > 0; s >>= 1) {
        if (t < s) sm[t] = fmaxf(sm[t], sm[t + s]);
        __syncthreads();
    }
    float r = sm[0]; __syncthreads();
    return r;
}

// ---------------------------------------------------------------------------
// K0: t_emb (f32) init + bf16 mirror
// ---------------------------------------------------------------------------
__global__ __launch_bounds__(256) void init_kernel(
    const float* __restrict__ emb, float* __restrict__ temb,
    unsigned short* __restrict__ temb_bf)
{
    size_t i = (size_t)blockIdx.x * 256 + threadIdx.x;
    float v = emb[i];
    temb[i] = v;
    temb_bf[i] = f2bf(v);
}

__global__ __launch_bounds__(256) void convw_kernel(
    const float* __restrict__ w, unsigned short* __restrict__ wbf)
{
    size_t i = (size_t)blockIdx.x * 256 + threadIdx.x;
    wbf[i] = f2bf(w[i]);
}

// ---------------------------------------------------------------------------
// K1: score[b,h,l] = relu( sum_d temb[b,l,d]*att_w[i,h,d] + att_b[i,h] )
//   WMMA bf16. Each wave owns one M-tile (16 rows of L) and computes ALL
//   four H-tiles with 4 accumulators -> A fragment loaded once per K-step.
//   All four B fragments are materialized in distinct registers BEFORE the
//   WMMA sequence so loads pipeline (no per-WMMA s_wait_loadcnt 0).
//   block = 128 (4 waves, consecutive M-tiles); grid = kB * (kL/64)
// ---------------------------------------------------------------------------
__global__ __launch_bounds__(128) void score_wmma_kernel(
    const unsigned short* __restrict__ temb_bf,
    const unsigned short* __restrict__ attw_bf,
    const float* __restrict__ att_b,
    float* __restrict__ score, int layer)
{
    int mtg = blockIdx.x & (kL / 64 - 1);   // 0..31
    int b   = blockIdx.x >> 5;
    int wave = threadIdx.x >> 5;
    int lane = threadIdx.x & 31;
    int half = lane >> 4;
    int ln   = lane & 15;
    int mt   = mtg * 4 + wave;              // M-tile of L

    const unsigned short* Arow = temb_bf + ((size_t)b * kL + mt * 16 + ln) * kD;
    const unsigned short* Brow = attw_bf + ((size_t)layer * kH + ln) * kD;  // + j*16*kD

    v8f acc[4];
    #pragma unroll
    for (int j = 0; j < 4; ++j) acc[j] = v8f{0.f,0.f,0.f,0.f,0.f,0.f,0.f,0.f};

    for (int k0 = 0; k0 < kD; k0 += 32) {
        Frag a, bb[4];
        a.q[0] = *(const uint4*)(Arow + k0 +      8 * half);
        a.q[1] = *(const uint4*)(Arow + k0 + 16 + 8 * half);
        __builtin_prefetch(Arow + k0 + 128, 0, 3);
        #pragma unroll
        for (int j = 0; j < 4; ++j) {
            const unsigned short* br = Brow + (size_t)j * 16 * kD;
            bb[j].q[0] = *(const uint4*)(br + k0 +      8 * half);
            bb[j].q[1] = *(const uint4*)(br + k0 + 16 + 8 * half);
        }
        #pragma unroll
        for (int j = 0; j < 4; ++j) {
            acc[j] = __builtin_amdgcn_wmma_f32_16x16x32_bf16(
                false, a.v, false, bb[j].v, (short)0, acc[j], false, false);
        }
    }

    #pragma unroll
    for (int j = 0; j < 4; ++j) {
        int h = j * 16 + ln;                // N index
        float bias = att_b[layer * kH + h];
        float* dst = score + ((size_t)b * kH + h) * kL + mt * 16 + 8 * half;
        float4 o0, o1;
        o0.x = fmaxf(acc[j][0] + bias, 0.f); o0.y = fmaxf(acc[j][1] + bias, 0.f);
        o0.z = fmaxf(acc[j][2] + bias, 0.f); o0.w = fmaxf(acc[j][3] + bias, 0.f);
        o1.x = fmaxf(acc[j][4] + bias, 0.f); o1.y = fmaxf(acc[j][5] + bias, 0.f);
        o1.z = fmaxf(acc[j][6] + bias, 0.f); o1.w = fmaxf(acc[j][7] + bias, 0.f);
        *(float4*)(dst)     = o0;
        *(float4*)(dst + 4) = o1;
    }
}

// ---------------------------------------------------------------------------
// K2: softmax over L per (b,h); writes att f32 (in place), att bf16,
//     and total_attention output slice.  grid = kB*kH, block = 256
// ---------------------------------------------------------------------------
__global__ __launch_bounds__(256) void softmax_kernel(
    float* __restrict__ att, unsigned short* __restrict__ att_bf,
    float* __restrict__ out_ta, int layer)
{
    __shared__ float sm[256];
    int bh = blockIdx.x;
    int t  = threadIdx.x;
    float* row = att + (size_t)bh * kL;

    float loc[8];
    float mx = -1e30f;
    #pragma unroll
    for (int j = 0; j < 8; ++j) { loc[j] = row[t + j * 256]; mx = fmaxf(mx, loc[j]); }
    mx = blockReduceMax(mx, sm);
    float s = 0.f;
    #pragma unroll
    for (int j = 0; j < 8; ++j) { loc[j] = expf(loc[j] - mx); s += loc[j]; }
    s = blockReduceSum(s, sm);
    float inv = 1.0f / s;

    float* ta = out_ta + ((size_t)layer * kB * kH + bh) * kL;
    #pragma unroll
    for (int j = 0; j < 8; ++j) {
        int l = t + j * 256;
        float a = loc[j] * inv;
        row[l] = a;
        att_bf[(size_t)bh * kL + l] = f2bf(a);
        ta[l] = a;
    }
}

// ---------------------------------------------------------------------------
// K3: att_sum[b,l] = sum_h att[b,h,l];  also attention_dis output
//     grid = kB*kL/256, block = 256
// ---------------------------------------------------------------------------
__global__ __launch_bounds__(256) void attsum_kernel(
    const float* __restrict__ att, float* __restrict__ att_sum,
    float* __restrict__ out_ad, int layer)
{
    int idx = blockIdx.x * 256 + threadIdx.x;     // b*kL + l
    int b = idx / kL;
    int l = idx - b * kL;
    float s = 0.f;
    #pragma unroll 4
    for (int h = 0; h < kH; ++h)
        s += att[((size_t)b * kH + h) * kL + l];
    att_sum[idx] = s;
    out_ad[(size_t)layer * kB * kL + idx] = s;
}

// ---------------------------------------------------------------------------
// K4: s[b,h,d] = (1/sqrt(L)) * sum_l att[b,h,l] * temb[b,l,d]
//   WMMA bf16: A = att (H x L, K-major rows). B = temb (L x D) needs
//   per-column access: stage 32(K) x 64(N) bf16 per wave through LDS
//   (wave-private -> no barriers). Each wave computes 4 N-tiles with a
//   single A fragment per K-step; all 4 B fragments gathered before the
//   WMMA sequence. Async global->LDS copies (ASYNCcnt) when available.
//   block = 128 (4 waves); grid = kB * 4(mt) * 4(256-col groups)
// ---------------------------------------------------------------------------
__global__ __launch_bounds__(128) void s_wmma_kernel(
    const unsigned short* __restrict__ att_bf,
    const unsigned short* __restrict__ temb_bf,
    float* __restrict__ sbuf)
{
    __shared__ __align__(16) unsigned short tile[4][4][32 * 16];  // [wave][j][k*16+n]

    int blk  = blockIdx.x;                  // kB * 16
    int b    = blk >> 4;
    int rem  = blk & 15;
    int mt   = rem >> 2;                    // H tile 0..3
    int ntg2 = rem & 3;                     // 256-column group
    int wave = threadIdx.x >> 5;
    int lane = threadIdx.x & 31;
    int half = lane >> 4;
    int ln   = lane & 15;
    int ntg  = ntg2 * 4 + wave;             // 64-column group, 0..15

    const unsigned short* Arow  = att_bf + ((size_t)b * kH + mt * 16 + ln) * kL;
    const unsigned short* Bbase = temb_bf + (size_t)b * kL * kD + (size_t)ntg * 64;

    v8f acc[4];
    #pragma unroll
    for (int j = 0; j < 4; ++j) acc[j] = v8f{0.f,0.f,0.f,0.f,0.f,0.f,0.f,0.f};

    for (int k0 = 0; k0 < kL; k0 += 32) {
        const unsigned short* src = Bbase + (size_t)(k0 + lane) * kD;  // row k0+lane, 64 cols
#ifdef USE_ASYNC_LDS
        #pragma unroll
        for (int j = 0; j < 4; ++j) {
            __builtin_amdgcn_global_load_async_to_lds_b128(
                (gv4_p)(src + j * 16),
                (lv4_p)&tile[wave][j][lane * 16],
                0, 0);
            __builtin_amdgcn_global_load_async_to_lds_b128(
                (gv4_p)(src + j * 16 + 8),
                (lv4_p)&tile[wave][j][lane * 16 + 8],
                0, 0);
        }
        wait_async_lds();
#else
        #pragma unroll
        for (int j = 0; j < 4; ++j) {
            uint4 r0 = *(const uint4*)(src + j * 16);
            uint4 r1 = *(const uint4*)(src + j * 16 + 8);
            *(uint4*)(&tile[wave][j][lane * 16])     = r0;
            *(uint4*)(&tile[wave][j][lane * 16 + 8]) = r1;
        }
#endif
        Frag a, bb[4];
        a.q[0] = *(const uint4*)(Arow + k0 +      8 * half);
        a.q[1] = *(const uint4*)(Arow + k0 + 16 + 8 * half);
        #pragma unroll
        for (int j = 0; j < 4; ++j) {
            #pragma unroll
            for (int i = 0; i < 8; ++i) {
                bb[j].s[i]     = tile[wave][j][(8 * half + i) * 16 + ln];
                bb[j].s[i + 8] = tile[wave][j][(16 + 8 * half + i) * 16 + ln];
            }
        }
        #pragma unroll
        for (int j = 0; j < 4; ++j) {
            acc[j] = __builtin_amdgcn_wmma_f32_16x16x32_bf16(
                false, a.v, false, bb[j].v, (short)0, acc[j], false, false);
        }
    }

    #pragma unroll
    for (int j = 0; j < 4; ++j) {
        int d = (ntg * 4 + j) * 16 + ln;    // N index
        #pragma unroll
        for (int v = 0; v < 8; ++v) {
            int h = mt * 16 + v + 8 * half; // M index
            sbuf[((size_t)b * kH + h) * kD + d] = acc[j][v] * kInvSqrtL;
        }
    }
}

// ---------------------------------------------------------------------------
// K5: layernorm(s) over D with att_ln, then per-head projection
//     vec[b,h,e] = relu( sum_d sn[d] * pro_w[i,h,e,d] + pro_b[i,h,e] )
//     stored permuted: out0[b][e*H + h].   grid = kB*kH, block = 256
// ---------------------------------------------------------------------------
__global__ __launch_bounds__(256) void ln_vec_kernel(
    const float* __restrict__ sbuf,
    const float* __restrict__ lng, const float* __restrict__ lnb,
    const float* __restrict__ prow, const float* __restrict__ prob,
    float* __restrict__ out0, int layer)
{
    __shared__ float sm[256];
    __shared__ float sn[kD];
    int bh = blockIdx.x;
    int b = bh / kH, h = bh - b * kH;
    int t = threadIdx.x;
    const float* row = sbuf + (size_t)bh * kD;

    float loc[4];
    float s = 0.f;
    #pragma unroll
    for (int j = 0; j < 4; ++j) { loc[j] = row[t + j * 256]; s += loc[j]; }
    float mu = blockReduceSum(s, sm) * (1.0f / kD);
    float vs = 0.f;
    #pragma unroll
    for (int j = 0; j < 4; ++j) { float d = loc[j] - mu; vs += d * d; }
    float var = blockReduceSum(vs, sm) * (1.0f / kD);
    float inv = rsqrtf(var + kEps);

    size_t gofs = ((size_t)layer * kH + h) * kD;
    #pragma unroll
    for (int j = 0; j < 4; ++j) {
        int d = t + j * 256;
        sn[d] = (loc[j] - mu) * inv * lng[gofs + d] + lnb[gofs + d];
    }
    __syncthreads();

    for (int e = 0; e < kE; ++e) {
        const float* w = prow + (((size_t)layer * kH + h) * kE + e) * kD;
        float p = 0.f;
        #pragma unroll
        for (int j = 0; j < 4; ++j) p += sn[t + j * 256] * w[t + j * 256];
        float tot = blockReduceSum(p, sm);
        if (t == 0) {
            float v = tot + prob[((size_t)layer * kH + h) * kE + e];
            out0[(size_t)b * kD + e * kH + h] = fmaxf(v, 0.f);
        }
    }
}

// ---------------------------------------------------------------------------
// K6: t_emb = LN( t_emb * (att_sum*proj_w + proj_b) + t_emb ) over D
//     also refresh bf16 mirror.   grid = kB*kL, block = 256
// ---------------------------------------------------------------------------
__global__ __launch_bounds__(256) void temb_update_kernel(
    float* __restrict__ temb, unsigned short* __restrict__ temb_bf,
    const float* __restrict__ att_sum,
    const float* __restrict__ projw, const float* __restrict__ projb,
    const float* __restrict__ ng, const float* __restrict__ nb, int layer)
{
    __shared__ float sm[256];
    int bl = blockIdx.x;
    int t  = threadIdx.x;
    float as = att_sum[bl];
    size_t base = (size_t)bl * kD;

    float loc[4];
    float s = 0.f;
    #pragma unroll
    for (int j = 0; j < 4; ++j) {
        int d = t + j * 256;
        float te = temb[base + d];
        float x = te * (as * projw[layer * kD + d] + projb[layer * kD + d]) + te;
        loc[j] = x; s += x;
    }
    float mu = blockReduceSum(s, sm) * (1.0f / kD);
    float vs = 0.f;
    #pragma unroll
    for (int j = 0; j < 4; ++j) { float d = loc[j] - mu; vs += d * d; }
    float var = blockReduceSum(vs, sm) * (1.0f / kD);
    float inv = rsqrtf(var + kEps);
    #pragma unroll
    for (int j = 0; j < 4; ++j) {
        int d = t + j * 256;
        float y = (loc[j] - mu) * inv * ng[layer * kD + d] + nb[layer * kD + d];
        temb[base + d] = y;
        temb_bf[base + d] = f2bf(y);
    }
}

// ---------------------------------------------------------------------------
// K7: out1[b,d] = relu( out0[b,:] . fi_w[i,d,:] + fi_b[i,d] ) + out0[b,d]
//     grid = kB*kD/256
// ---------------------------------------------------------------------------
__global__ __launch_bounds__(256) void fi_kernel(
    const float* __restrict__ out0, const float* __restrict__ fi_w,
    const float* __restrict__ fi_b, float* __restrict__ out1, int layer)
{
    int idx = blockIdx.x * 256 + threadIdx.x;   // b*kD + d
    int b = idx >> 10;
    int d = idx & (kD - 1);
    const float* x = out0 + (size_t)b * kD;
    const float* w = fi_w + ((size_t)layer * kD + d) * kD;
    float acc = fi_b[layer * kD + d];
    #pragma unroll 4
    for (int k = 0; k < kD; ++k) acc += x[k] * w[k];
    out1[idx] = fmaxf(acc, 0.f) + x[d];
}

// ---------------------------------------------------------------------------
// K8: ott[b, i*S+s] = relu( out1[b,:] . set_w[i,s,:] + set_b[i,s] )
//     grid = kB*kS/256
// ---------------------------------------------------------------------------
__global__ __launch_bounds__(256) void set_kernel(
    const float* __restrict__ out1, const float* __restrict__ set_w,
    const float* __restrict__ set_b, float* __restrict__ ott, int layer)
{
    int idx = blockIdx.x * 256 + threadIdx.x;   // b*kS + s
    int b = idx >> 8;
    int s = idx & (kS - 1);
    const float* x = out1 + (size_t)b * kD;
    const float* w = set_w + ((size_t)layer * kS + s) * kD;
    float acc = set_b[layer * kS + s];
    #pragma unroll 4
    for (int k = 0; k < kD; ++k) acc += x[k] * w[k];
    ott[(size_t)b * (kNL * kS) + layer * kS + s] = fmaxf(acc, 0.f);
}

// ---------------------------------------------------------------------------
// K9: fsc[b,l] = relu( t_emb[b,l,:] . aat_w + aat_b )   grid = kB*kL
// ---------------------------------------------------------------------------
__global__ __launch_bounds__(256) void final_score_kernel(
    const float* __restrict__ temb, const float* __restrict__ aat_w,
    const float* __restrict__ aat_b, float* __restrict__ fsc)
{
    __shared__ float sm[256];
    int bl = blockIdx.x;
    int t  = threadIdx.x;
    const float* row = temb + (size_t)bl * kD;
    float s = 0.f;
    #pragma unroll
    for (int j = 0; j < 4; ++j) s += row[t + j * 256] * aat_w[t + j * 256];
    s = blockReduceSum(s, sm);
    if (t == 0) fsc[bl] = fmaxf(s + aat_b[0], 0.f);
}

// ---------------------------------------------------------------------------
// K10: final softmax over L per b; writes fatt + final_att output. grid = kB
// ---------------------------------------------------------------------------
__global__ __launch_bounds__(256) void final_softmax_kernel(
    const float* __restrict__ fsc, float* __restrict__ fatt,
    float* __restrict__ out_fa)
{
    __shared__ float sm[256];
    int b = blockIdx.x;
    int t = threadIdx.x;
    const float* row = fsc + (size_t)b * kL;
    float loc[8];
    float mx = -1e30f;
    #pragma unroll
    for (int j = 0; j < 8; ++j) { loc[j] = row[t + j * 256]; mx = fmaxf(mx, loc[j]); }
    mx = blockReduceMax(mx, sm);
    float s = 0.f;
    #pragma unroll
    for (int j = 0; j < 8; ++j) { loc[j] = expf(loc[j] - mx); s += loc[j]; }
    s = blockReduceSum(s, sm);
    float inv = 1.0f / s;
    #pragma unroll
    for (int j = 0; j < 8; ++j) {
        int l = t + j * 256;
        float a = loc[j] * inv;
        fatt[(size_t)b * kL + l] = a;
        out_fa[(size_t)b * kL + l] = a;
    }
}

// ---------------------------------------------------------------------------
// K11: sum_[b,d] = LN( (1/sqrt(L)) * sum_l temb[b,l,d]*fatt[b,l] )  grid = kB
// ---------------------------------------------------------------------------
__global__ __launch_bounds__(256) void final_sum_ln_kernel(
    const float* __restrict__ temb, const float* __restrict__ fatt,
    const float* __restrict__ g, const float* __restrict__ bb,
    float* __restrict__ sumln)
{
    __shared__ float sm[256];
    __shared__ float sv[kD];
    __shared__ float fa[kL];
    int b = blockIdx.x;
    int t = threadIdx.x;
    for (int l = t; l < kL; l += 256) fa[l] = fatt[(size_t)b * kL + l];
    __syncthreads();

    float acc[4] = {0.f, 0.f, 0.f, 0.f};
    for (int l = 0; l < kL; ++l) {
        float w = fa[l];
        const float* row = temb + ((size_t)b * kL + l) * kD;
        #pragma unroll
        for (int j = 0; j < 4; ++j) acc[j] += row[t + j * 256] * w;
    }
    float s = 0.f;
    #pragma unroll
    for (int j = 0; j < 4; ++j) { acc[j] *= kInvSqrtL; s += acc[j]; sv[t + j * 256] = acc[j]; }
    float mu = blockReduceSum(s, sm) * (1.0f / kD);
    float vs = 0.f;
    #pragma unroll
    for (int j = 0; j < 4; ++j) { float d = acc[j] - mu; vs += d * d; }
    float var = blockReduceSum(vs, sm) * (1.0f / kD);
    float inv = rsqrtf(var + kEps);
    #pragma unroll
    for (int j = 0; j < 4; ++j) {
        int d = t + j * 256;
        sumln[(size_t)b * kD + d] = (sv[d] - mu) * inv * g[d] + bb[d];
    }
}

// ---------------------------------------------------------------------------
// K12: P[b] = sigmoid( concat(sumln[b], ott[b]) . h2p_w + h2p_b )  grid = kB
// ---------------------------------------------------------------------------
__global__ __launch_bounds__(256) void head_kernel(
    const float* __restrict__ sumln, const float* __restrict__ ott,
    const float* __restrict__ w, const float* __restrict__ bias,
    float* __restrict__ outP)
{
    __shared__ float sm[256];
    int b = blockIdx.x;
    int t = threadIdx.x;
    float s = 0.f;
    for (int c = t; c < kCAT; c += 256) {
        float x = (c < kD) ? sumln[(size_t)b * kD + c]
                           : ott[(size_t)b * (kNL * kS) + (c - kD)];
        s += x * w[c];
    }
    s = blockReduceSum(s, sm);
    if (t == 0) outP[b] = 1.0f / (1.0f + expf(-(s + bias[0])));
}

// ---------------------------------------------------------------------------
// Host launch
// ---------------------------------------------------------------------------
extern "C" void kernel_launch(void* const* d_in, const int* in_sizes, int n_in,
                              void* d_out, int out_size, void* d_ws, size_t ws_size,
                              hipStream_t stream)
{
    const float* emb      = (const float*)d_in[0];
    const float* att_w    = (const float*)d_in[1];
    const float* att_b    = (const float*)d_in[2];
    const float* att_ln_g = (const float*)d_in[3];
    const float* att_ln_b = (const float*)d_in[4];
    const float* pro_w    = (const float*)d_in[5];
    const float* pro_b    = (const float*)d_in[6];
    const float* proj_w   = (const float*)d_in[7];
    const float* proj_b   = (const float*)d_in[8];
    const float* fi_w     = (const float*)d_in[9];
    const float* fi_b     = (const float*)d_in[10];
    const float* set_w    = (const float*)d_in[11];
    const float* set_b    = (const float*)d_in[12];
    const float* norm_g   = (const float*)d_in[13];
    const float* norm_b   = (const float*)d_in[14];
    const float* aat_w    = (const float*)d_in[15];
    const float* aat_b    = (const float*)d_in[16];
    const float* aat_ln_g = (const float*)d_in[17];
    const float* aat_ln_b = (const float*)d_in[18];
    const float* h2p_w    = (const float*)d_in[19];
    const float* h2p_b    = (const float*)d_in[20];

    // output layout: P[8] | attention_dis[NL*B*L] | total_attention[NL*B*H*L] | final_att[B*L]
    float* out    = (float*)d_out;
    float* out_P  = out;
    float* out_ad = out + 8;
    float* out_ta = out_ad + (size_t)kNL * kB * kL;
    float* out_fa = out_ta + (size_t)kNL * kB * kH * kL;

    // workspace carve-up (all chunks 256B-aligned sizes)
    char* ws = (char*)d_ws;
    float* temb    = (float*)ws;                 ws += (size_t)kB * kL * kD * 4;   // 64 MB
    float* score   = (float*)ws;                 ws += (size_t)kB * kH * kL * 4;   // 4 MB
    float* sbuf    = (float*)ws;                 ws += (size_t)kB * kH * kD * 4;   // 2 MB
    float* att_sum = (float*)ws;                 ws += (size_t)kB * kL * 4;
    float* out0    = (float*)ws;                 ws += (size_t)kB * kD * 4;
    float* out1    = (float*)ws;                 ws += (size_t)kB * kD * 4;
    float* ott     = (float*)ws;                 ws += (size_t)kB * kNL * kS * 4;
    float* fsc     = (float*)ws;                 ws += (size_t)kB * kL * 4;
    float* fatt    = (float*)ws;                 ws += (size_t)kB * kL * 4;
    float* sumln   = (float*)ws;                 ws += (size_t)kB * kD * 4;
    unsigned short* temb_bf = (unsigned short*)ws; ws += (size_t)kB * kL * kD * 2; // 32 MB
    unsigned short* attw_bf = (unsigned short*)ws; ws += (size_t)kNL * kH * kD * 2;
    unsigned short* att_bf  = (unsigned short*)ws; ws += (size_t)kB * kH * kL * 2;

    // init: t_emb copy + bf16 mirror; bf16 attention weights
    init_kernel<<<(kB * kL * kD) / 256, 256, 0, stream>>>(emb, temb, temb_bf);
    convw_kernel<<<(kNL * kH * kD) / 256, 256, 0, stream>>>(att_w, attw_bf);

    for (int i = 0; i < kNL; ++i) {
        score_wmma_kernel<<<kB * (kL / 64), 128, 0, stream>>>(
            temb_bf, attw_bf, att_b, score, i);
        softmax_kernel<<<kB * kH, 256, 0, stream>>>(score, att_bf, out_ta, i);
        attsum_kernel<<<(kB * kL) / 256, 256, 0, stream>>>(score, att_sum, out_ad, i);
        s_wmma_kernel<<<kB * 16, 128, 0, stream>>>(att_bf, temb_bf, sbuf);
        ln_vec_kernel<<<kB * kH, 256, 0, stream>>>(
            sbuf, att_ln_g, att_ln_b, pro_w, pro_b, out0, i);
        temb_update_kernel<<<kB * kL, 256, 0, stream>>>(
            temb, temb_bf, att_sum, proj_w, proj_b, norm_g, norm_b, i);
        fi_kernel<<<(kB * kD) / 256, 256, 0, stream>>>(out0, fi_w, fi_b, out1, i);
        set_kernel<<<(kB * kS) / 256, 256, 0, stream>>>(out1, set_w, set_b, ott, i);
    }

    final_score_kernel<<<kB * kL, 256, 0, stream>>>(temb, aat_w, aat_b, fsc);
    final_softmax_kernel<<<kB, 256, 0, stream>>>(fsc, fatt, out_fa);
    final_sum_ln_kernel<<<kB, 256, 0, stream>>>(temb, fatt, aat_ln_g, aat_ln_b, sumln);
    head_kernel<<<kB, 256, 0, stream>>>(sumln, ott, h2p_w, h2p_b, out_P);
}
